// MeanPool_29540785062224
// MI455X (gfx1250) — matrix operand order repaired and checked by reading, not compiled, not run on hardware
//
#include <hip/hip_runtime.h>

typedef __attribute__((ext_vector_type(16))) _Float16 v16h;
typedef __attribute__((ext_vector_type(8)))  float    v8f;

#define TS 72          // LDS tile row stride in halves (pad to dodge bank conflicts)
#define NEG 0.01f      // LeakyReLU slope
#define LN_EPS 1e-5f

#define WMMA(a, bf, c) __builtin_amdgcn_wmma_f32_16x16x32_f16(false, (a), false, (bf), (short)0, (c), false, false)

__device__ __forceinline__ float lrelu(float v) { return v > 0.f ? v : NEG * v; }

// ---- A fragment (16x32 f16), K chunk kc, from row-major LDS tile (stride TS halves) ----
// ISA layout: lane<16 row M=lane holds K in {0..7,16..23}; lanes 16..31 hold K+8.
__device__ __forceinline__ v16h a_frag(const _Float16* t, int lane, int kc) {
  union { v16h v; unsigned u[8]; } r;
  const int m  = lane & 15;
  const int kb = ((lane >> 4) << 3) + (kc << 5);
  const _Float16* row = t + m * TS;
  #pragma unroll
  for (int p = 0; p < 8; ++p) {
    const int k = kb + ((2 * p) & 7) + ((p >> 2) << 4); // pairs {0..7} then {16..23}
    r.u[p] = *(const unsigned*)(row + k);
  }
  return r.v;
}

// ---- B fragment (32x16): B[K][n] = W[n][k]; W row-major [outDim][64] fp32 in global ----
// ISA layout: lane = column n (mod 16), elems hold K in {0..7,16..23}, lanes 16..31 get K+8.
__device__ __forceinline__ v16h b_frag(const float* __restrict__ W, int lane, int nc, int kc) {
  v16h r;
  const int n  = (nc << 4) + (lane & 15);
  const int k0 = (kc << 5) + ((lane >> 4) << 3);
  const float* row = W + n * 64 + k0;
  #pragma unroll
  for (int e = 0; e < 16; ++e) r[e] = (_Float16)row[(e & 7) + ((e >> 3) << 4)];
  return r;
}

// ---- In-place LayerNorm over 64 cols of a 16-row f16 LDS tile; 2 lanes per row ----
__device__ __forceinline__ void ln_tile(_Float16* t, int lane,
                                        const float* __restrict__ w,
                                        const float* __restrict__ b) {
  const int r = lane >> 1, h = lane & 1;
  _Float16* p = t + r * TS + (h << 5);
  float x[32], s = 0.f, ss = 0.f;
  #pragma unroll
  for (int j = 0; j < 32; ++j) { float v = (float)p[j]; x[j] = v; s += v; ss += v * v; }
  s  += __shfl_xor(s, 1, 32);
  ss += __shfl_xor(ss, 1, 32);
  const float mu   = s * 0.015625f;
  const float var  = ss * 0.015625f - mu * mu;
  const float rstd = rsqrtf(var + LN_EPS);
  #pragma unroll
  for (int j = 0; j < 32; ++j) {
    const int c = (h << 5) + j;
    p[j] = (_Float16)((x[j] - mu) * rstd * w[c] + b[c]);
  }
}

// ---- Fill a 16x64 LDS f16 tile from a row-major fp32 matrix (rows >= nrows -> 0) ----
__device__ __forceinline__ void fill_tile(_Float16* t, int lane,
                                          const float* __restrict__ src,
                                          long base, long nrows) {
  const int r = lane >> 1, h = lane & 1;
  const long row = base + r;
  _Float16* p = t + r * TS + (h << 5);
  if (row < nrows) {
    const float4* g = (const float4*)(src + row * 64 + (h << 5));
    #pragma unroll
    for (int q = 0; q < 8; ++q) {
      float4 v = g[q];
      p[q * 4 + 0] = (_Float16)v.x; p[q * 4 + 1] = (_Float16)v.y;
      p[q * 4 + 2] = (_Float16)v.z; p[q * 4 + 3] = (_Float16)v.w;
    }
  } else {
    #pragma unroll
    for (int j = 0; j < 32; ++j) p[j] = (_Float16)0.f;
  }
}

// =====================  node kernel: hn + emb_local (accumulator init)  =====================
__global__ __launch_bounds__(256) void node_kernel(
    const float* __restrict__ node_z,
    const float* __restrict__ n_ln_w, const float* __restrict__ n_ln_b,
    const float* __restrict__ node_W, const float* __restrict__ node_b,
    const float* __restrict__ r_ln_w, const float* __restrict__ r_ln_b,
    const float* __restrict__ red_W,  const float* __restrict__ red_b,
    _Float16* __restrict__ hn, float* __restrict__ out, int* __restrict__ deg, long N) {
  __shared__ __align__(16) _Float16 tiles[8][16 * TS];
  const int lane = threadIdx.x & 31, wave = threadIdx.x >> 5;
  _Float16* tile = tiles[wave];
  const long waveBase = (long)blockIdx.x * 128 + wave * 16;

  { const long gid = (long)blockIdx.x * blockDim.x + threadIdx.x; if (gid < N) deg[gid] = 0; }

  fill_tile(tile, lane, node_z, waveBase, N);
  __syncthreads();
  ln_tile(tile, lane, n_ln_w, n_ln_b);
  __syncthreads();

  // hn = leaky(LN(node_z) @ node_W^T + node_b)   [16 x 32]
  v16h a0 = a_frag(tile, lane, 0), a1 = a_frag(tile, lane, 1);
  v8f hc[2];
  #pragma unroll
  for (int nc = 0; nc < 2; ++nc) {
    v8f acc = {};
    acc = WMMA(a0, b_frag(node_W, lane, nc, 0), acc);
    acc = WMMA(a1, b_frag(node_W, lane, nc, 1), acc);
    hc[nc] = acc;
  }
  __syncthreads();
  // tile := [hn | 0]
  #pragma unroll
  for (int nc = 0; nc < 2; ++nc) {
    const float bias = node_b[(nc << 4) + (lane & 15)];
    #pragma unroll
    for (int v = 0; v < 8; ++v) {
      const int m = v + ((lane >> 4) << 3);
      tile[m * TS + (nc << 4) + (lane & 15)] = (_Float16)lrelu(hc[nc][v] + bias);
    }
  }
  {
    const int r = lane >> 1, h = lane & 1;
    _Float16* z = tile + r * TS + 32 + (h << 4);
    #pragma unroll
    for (int j = 0; j < 16; ++j) z[j] = (_Float16)0.f;
  }
  __syncthreads();
  // persist hn (f16, 64B/node) for the edge-kernel gather (L2 resident: 6.4 MB)
  {
    const int r = lane >> 1, h = lane & 1;
    const long row = waveBase + r;
    if (row < N) {
      const uint4* t4 = (const uint4*)(tile + r * TS + (h << 4));
      uint4* g = (uint4*)(hn + row * 32 + (h << 4));
      g[0] = t4[0]; g[1] = t4[1];
    }
  }
  __syncthreads();
  // emb_local = leaky(LN([hn,0]) @ red_W^T + red_b) -> initializes out accumulator
  ln_tile(tile, lane, r_ln_w, r_ln_b);
  __syncthreads();
  a0 = a_frag(tile, lane, 0); a1 = a_frag(tile, lane, 1);
  #pragma unroll
  for (int nc = 0; nc < 4; ++nc) {
    v8f acc = {};
    acc = WMMA(a0, b_frag(red_W, lane, nc, 0), acc);
    acc = WMMA(a1, b_frag(red_W, lane, nc, 1), acc);
    const int n = (nc << 4) + (lane & 15);
    const float bias = red_b[n];
    #pragma unroll
    for (int v = 0; v < 8; ++v) {
      const int m = v + ((lane >> 4) << 3);
      const long row = waveBase + m;
      if (row < N) out[row * 64 + n] = lrelu(acc[v] + bias);
    }
  }
}

// =====================  edge kernel: he, gather, reduce MLP, scatter-add  =====================
__global__ __launch_bounds__(256) void edge_kernel(
    const float* __restrict__ edge_z,
    const int* __restrict__ src, const int* __restrict__ dst,
    const float* __restrict__ e_ln_w, const float* __restrict__ e_ln_b,
    const float* __restrict__ edge_W, const float* __restrict__ edge_b,
    const float* __restrict__ r_ln_w, const float* __restrict__ r_ln_b,
    const float* __restrict__ red_W,  const float* __restrict__ red_b,
    const _Float16* __restrict__ hn, float* __restrict__ out, int* __restrict__ deg, long E) {
  __shared__ __align__(16) _Float16 tiles[8][16 * TS];
  __shared__ unsigned dstb[8][16];
  const int lane = threadIdx.x & 31, wave = threadIdx.x >> 5;
  _Float16* tile = tiles[wave];
  const long waveBase = (long)blockIdx.x * 128 + wave * 16;

  fill_tile(tile, lane, edge_z, waveBase, E);
  __syncthreads();
  ln_tile(tile, lane, e_ln_w, e_ln_b);
  __syncthreads();

  // he = leaky(LN(edge_z) @ edge_W^T + edge_b)   [16 x 32]
  v16h a0 = a_frag(tile, lane, 0), a1 = a_frag(tile, lane, 1);
  v8f ec[2];
  #pragma unroll
  for (int nc = 0; nc < 2; ++nc) {
    v8f acc = {};
    acc = WMMA(a0, b_frag(edge_W, lane, nc, 0), acc);
    acc = WMMA(a1, b_frag(edge_W, lane, nc, 1), acc);
    ec[nc] = acc;
  }
  __syncthreads();
  // tile := [hn[src] | he]
  #pragma unroll
  for (int nc = 0; nc < 2; ++nc) {
    const float bias = edge_b[(nc << 4) + (lane & 15)];
    #pragma unroll
    for (int v = 0; v < 8; ++v) {
      const int m = v + ((lane >> 4) << 3);
      tile[m * TS + 32 + (nc << 4) + (lane & 15)] = (_Float16)lrelu(ec[nc][v] + bias);
    }
  }
  {
    const int r = lane >> 1, h = lane & 1;
    const long row = waveBase + r;
    const int sn = (row < E) ? src[row] : 0;
    const uint4* g = (const uint4*)(hn + (long)sn * 32);
    uint4* t4 = (uint4*)(tile + r * TS + (h << 4));
    t4[0] = g[2 * h]; t4[1] = g[2 * h + 1];
  }
  if (lane < 16) {
    const long row = waveBase + lane;
    dstb[wave][lane] = (row < E) ? (unsigned)dst[row] * 64u : 0u;
  }
  __syncthreads();
  ln_tile(tile, lane, r_ln_w, r_ln_b);
  __syncthreads();
  a0 = a_frag(tile, lane, 0); a1 = a_frag(tile, lane, 1);
  #pragma unroll
  for (int nc = 0; nc < 4; ++nc) {
    v8f acc = {};
    acc = WMMA(a0, b_frag(red_W, lane, nc, 0), acc);
    acc = WMMA(a1, b_frag(red_W, lane, nc, 1), acc);
    const int n = (nc << 4) + (lane & 15);
    const float bias = red_b[n];
    #pragma unroll
    for (int v = 0; v < 8; ++v) {
      const int m = v + ((lane >> 4) << 3);
      if (waveBase + m < E) atomicAdd(out + dstb[wave][m] + n, lrelu(acc[v] + bias));
    }
  }
  if (lane < 16) {
    const long row = waveBase + lane;
    if (row < E) atomicAdd(deg + dst[row], 1);
  }
}

// =====================  finalize: divide by (deg + 1)  =====================
__global__ __launch_bounds__(256) void fin_kernel(float* __restrict__ out,
                                                  const int* __restrict__ deg, long total) {
  const long i = (long)blockIdx.x * blockDim.x + threadIdx.x;
  if (i < total) out[i] = out[i] / (float)(deg[i >> 6] + 1);
}

extern "C" void kernel_launch(void* const* d_in, const int* in_sizes, int n_in,
                              void* d_out, int out_size, void* d_ws, size_t ws_size,
                              hipStream_t stream) {
  const float* node_z = (const float*)d_in[0];
  const float* edge_z = (const float*)d_in[1];
  const int*   src    = (const int*)d_in[2];
  const int*   dst    = (const int*)d_in[3];
  const float* n_ln_w = (const float*)d_in[4];
  const float* n_ln_b = (const float*)d_in[5];
  const float* node_W = (const float*)d_in[6];
  const float* node_b = (const float*)d_in[7];
  const float* e_ln_w = (const float*)d_in[8];
  const float* e_ln_b = (const float*)d_in[9];
  const float* edge_W = (const float*)d_in[10];
  const float* edge_b = (const float*)d_in[11];
  const float* r_ln_w = (const float*)d_in[12];
  const float* r_ln_b = (const float*)d_in[13];
  const float* red_W  = (const float*)d_in[14];
  const float* red_b  = (const float*)d_in[15];

  const long N = (long)in_sizes[0] / 64;
  const long E = (long)in_sizes[2];
  float* out = (float*)d_out;
  _Float16* hn = (_Float16*)d_ws;                       // N*32 f16 = N*64 bytes
  int* deg = (int*)((char*)d_ws + (size_t)N * 64);      // N ints

  const int nBlocksN = (int)((N + 127) / 128);
  const int nBlocksE = (int)((E + 127) / 128);
  node_kernel<<<nBlocksN, 256, 0, stream>>>(node_z, n_ln_w, n_ln_b, node_W, node_b,
                                            r_ln_w, r_ln_b, red_W, red_b,
                                            hn, out, deg, N);
  edge_kernel<<<nBlocksE, 256, 0, stream>>>(edge_z, src, dst, e_ln_w, e_ln_b, edge_W, edge_b,
                                            r_ln_w, r_ln_b, red_W, red_b,
                                            hn, out, deg, E);
  const long total = N * 64;
  fin_kernel<<<(int)((total + 255) / 256), 256, 0, stream>>>(out, deg, total);
}
